// MultiHeadAttention_18133351923719
// MI455X (gfx1250) — matrix operand order repaired and checked by reading, not compiled, not run on hardware
//
#include <hip/hip_runtime.h>

// ---------------- types matching gfx1250 WMMA builtins ----------------
typedef __attribute__((ext_vector_type(16))) __bf16 v16bf;
typedef __attribute__((ext_vector_type(8)))  __bf16 bf16x8;
typedef __attribute__((ext_vector_type(4)))  __bf16 bf16x4;
typedef __attribute__((ext_vector_type(8)))  float  v8f;
typedef __attribute__((ext_vector_type(4)))  float  f32x4;

#define T_SEQ 2048
#define BATCH 4
#define EMB   1024
#define HEADS 16
#define HD    64
#define MROWS (T_SEQ * BATCH)   // 8192

static __device__ __forceinline__ v8f wmma_bf16(v16bf a, v16bf b, v8f c) {
  return __builtin_amdgcn_wmma_f32_16x16x32_bf16(false, a, false, b,
                                                 (short)0, c, false, false);
}

// CDNA5 async copy: 16B global -> LDS, tracked by ASYNCcnt (no VGPR data path).
static __device__ __forceinline__ void async_cp16(void* lds, const void* g) {
  // generic LDS pointer: low 32 bits are the LDS byte offset (ISA 10.2 aperture rule)
  asm volatile("global_load_async_to_lds_b128 %0, %1, off"
               :: "v"((unsigned)(unsigned long long)lds), "v"(g)
               : "memory");
}
static __device__ __forceinline__ void wait_async() {
  asm volatile("s_wait_asynccnt 0x0" ::: "memory");
}

// A-fragment (16x32 bf16): lane holds row (lane&15); chunk0 = K[(lane>>4)*16 .. +7],
// chunk1 = K[(lane>>4)*16+8 .. +7]  (contiguous k order, matches B fragments)
static __device__ __forceinline__ v16bf load_afrag(const __bf16* rowbase, int lane) {
  union { v16bf v; bf16x8 h[2]; } u;
  const int o = (lane >> 4) * 16;
  u.h[0] = *(const bf16x8*)(rowbase + o);
  u.h[1] = *(const bf16x8*)(rowbase + o + 8);
  return u.v;
}

// ======================================================================
// f32 -> bf16 conversion pass (bandwidth-trivial; makes all GEMM staging
// a pure byte copy so it can ride GLOBAL_LOAD_ASYNC_TO_LDS)
// ======================================================================
__global__ __launch_bounds__(256) void cvt_bf16_kernel(const float* __restrict__ in,
                                                       __bf16* __restrict__ out,
                                                       int n4) {
  const int i = blockIdx.x * 256 + threadIdx.x;
  if (i < n4) {
    f32x4 x = ((const f32x4*)in)[i];
    bf16x4 y;
#pragma unroll
    for (int e = 0; e < 4; ++e) y[e] = (__bf16)x[e];
    ((bf16x4*)out)[i] = y;
  }
}

// ======================================================================
// Unified bf16 GEMM  Y = A @ W^T   (A: MROWSxEMB bf16, W: EMBxEMB bf16)
// mode 0 : bf16 out, head-major (b,h,t,d)          (Q, K projections)
// mode 1 : bf16 out, transposed head-major (b,h,d,t)  (V projection)
// mode 2 : f32 out, row-major MROWSxEMB            (final output proj)
// Double-buffered LDS tiles filled with async-to-LDS; 1 barrier / K-step.
// ======================================================================
__global__ __launch_bounds__(256) void gemm_kernel(const __bf16* __restrict__ A,
                                                   const __bf16* __restrict__ W,
                                                   __bf16* __restrict__ OutB,
                                                   float* __restrict__ OutF,
                                                   int mode) {
  __shared__ __align__(32) __bf16 lA[2][128][32];
  __shared__ __align__(32) __bf16 lB[2][128][32];
  const int tid  = threadIdx.x;
  const int lane = tid & 31, wave = tid >> 5;
  const int Mbase = blockIdx.y * 128;
  const int Nbase = blockIdx.x * 128;

  const int srow = tid >> 1;            // staging: 0..127
  const int skq  = (tid & 1) * 16;      // 0 or 16 bf16 (32B per thread per matrix)
  const __bf16* Ab = A + (size_t)(Mbase + srow) * EMB + skq;
  const __bf16* Wb = W + (size_t)(Nbase + srow) * EMB + skq;

  // prologue: stage K-chunk 0 into buffer 0
  async_cp16(&lA[0][srow][skq], Ab);
  async_cp16(&lA[0][srow][skq + 8], Ab + 8);
  async_cp16(&lB[0][srow][skq], Wb);
  async_cp16(&lB[0][srow][skq + 8], Wb + 8);

  v8f acc[8];
#pragma unroll
  for (int s = 0; s < 8; ++s) acc[s] = (v8f){0.f,0.f,0.f,0.f,0.f,0.f,0.f,0.f};

  for (int kk = 0; kk < EMB; kk += 32) {
    const int cur = (kk >> 5) & 1;
    wait_async();
    __syncthreads();                    // all waves' tiles for `cur` landed
    if (kk + 32 < EMB) {                // kick next chunk into other buffer
      const int nxt = cur ^ 1;
      async_cp16(&lA[nxt][srow][skq], Ab + kk + 32);
      async_cp16(&lA[nxt][srow][skq + 8], Ab + kk + 40);
      async_cp16(&lB[nxt][srow][skq], Wb + kk + 32);
      async_cp16(&lB[nxt][srow][skq + 8], Wb + kk + 40);
    }
    const v16bf a = load_afrag(&lA[cur][wave * 16 + (lane & 15)][0], lane);
#pragma unroll
    for (int s = 0; s < 8; ++s) {
      const v16bf b = *(const v16bf*)&lB[cur][s * 16 + (lane & 15)][(lane >> 4) * 16];
      acc[s] = wmma_bf16(a, b, acc[s]);
    }
  }

  // ---- epilogue: scatter per mode ----
#pragma unroll
  for (int s = 0; s < 8; ++s) {
    const int f = Nbase + s * 16 + (lane & 15);
    const int h = f >> 6, d = f & 63;
#pragma unroll
    for (int i = 0; i < 8; ++i) {
      const int r = Mbase + wave * 16 + i + ((lane >> 4) << 3);
      if (mode == 2) {
        OutF[(size_t)r * EMB + f] = acc[s][i];
      } else {
        const int t = r >> 2, b = r & 3;
        size_t idx;
        if (mode == 1) idx = ((size_t)(b * HEADS + h) * HD + d) * T_SEQ + t;
        else           idx = ((size_t)(b * HEADS + h) * T_SEQ + t) * HD + d;
        OutB[idx] = (__bf16)acc[s][i];
      }
    }
  }
}

// ======================================================================
// Flash attention.  grid = (Tq/128, B*H), 8 waves x 16 q-rows.
// K / V^T blocks double-buffered via async-to-LDS; 1 barrier / 64-key block.
// ======================================================================
__global__ __launch_bounds__(256) void flash_kernel(const __bf16* __restrict__ Qh,
                                                    const __bf16* __restrict__ Kh,
                                                    const __bf16* __restrict__ Vt,
                                                    const unsigned char* __restrict__ mask,
                                                    __bf16* __restrict__ An) {
  __shared__ __align__(32) __bf16 lK[2][64][64];   // [kt][d]
  __shared__ __align__(32) __bf16 lV[2][64][64];   // [d][kt]
  __shared__ __align__(32) __bf16 lP[8][16][64];   // per-wave P tile
  __shared__ unsigned long long   lM[256];         // key-padding mask for this b

  const int tid = threadIdx.x, lane = tid & 31, wave = tid >> 5;
  const int bh = blockIdx.y;
  const int b = bh >> 4, h = bh & 15;
  const int q0 = blockIdx.x * 128 + wave * 16;

  const int srow = tid >> 2;            // staging: 0..63
  const int scol = (tid & 3) * 16;      // 32B per thread per matrix
  const __bf16* Kb = Kh + ((size_t)bh * T_SEQ + srow) * HD + scol;
  const __bf16* Vb = Vt + ((size_t)bh * HD + srow) * T_SEQ + scol;

  lM[tid] = ((const unsigned long long*)(mask + (size_t)b * T_SEQ))[tid];
  const unsigned char* mrow = (const unsigned char*)lM;

  // prologue: stage key-block 0
  async_cp16(&lK[0][srow][scol], Kb);
  async_cp16(&lK[0][srow][scol + 8], Kb + 8);
  async_cp16(&lV[0][srow][scol], Vb);
  async_cp16(&lV[0][srow][scol + 8], Vb + 8);

  // Q fragments for this wave's 16 rows, kept in registers all kernel
  const __bf16* Qb = Qh + ((size_t)bh * T_SEQ + q0) * HD;
  v16bf qa[2];
#pragma unroll
  for (int c = 0; c < 2; ++c)
    qa[c] = load_afrag(Qb + (size_t)(lane & 15) * HD + c * 32, lane);

  float mi[8], li[8];
  v8f oacc[4];
#pragma unroll
  for (int i = 0; i < 8; ++i) { mi[i] = -1e30f; li[i] = 0.f; }
#pragma unroll
  for (int j = 0; j < 4; ++j) oacc[j] = (v8f){0.f,0.f,0.f,0.f,0.f,0.f,0.f,0.f};

  for (int kb = 0; kb < T_SEQ; kb += 64) {
    const int cur = (kb >> 6) & 1;
    wait_async();
    __syncthreads();                   // this block's K/V landed (+ mask, 1st iter)
    if (kb + 64 < T_SEQ) {             // prefetch next key-block into other buffer
      const int nxt = cur ^ 1;
      const int ko = (kb + 64);
      async_cp16(&lK[nxt][srow][scol], Kb + (size_t)ko * HD);
      async_cp16(&lK[nxt][srow][scol + 8], Kb + (size_t)ko * HD + 8);
      async_cp16(&lV[nxt][srow][scol], Vb + ko);
      async_cp16(&lV[nxt][srow][scol + 8], Vb + ko + 8);
    }

    // ---- S = Q K^T / 8, with key-padding mask ----
    v8f s4[4];
#pragma unroll
    for (int j = 0; j < 4; ++j) {
      v8f s = (v8f){0.f,0.f,0.f,0.f,0.f,0.f,0.f,0.f};
#pragma unroll
      for (int c = 0; c < 2; ++c) {
        const v16bf kf = *(const v16bf*)&lK[cur][j * 16 + (lane & 15)][c * 32 + (lane >> 4) * 16];
        s = wmma_bf16(qa[c], kf, s);
      }
      const int ktg = kb + j * 16 + (lane & 15);   // this lane's key column
      if (mrow[ktg]) {
#pragma unroll
        for (int i = 0; i < 8; ++i) s[i] = -1e9f;
      } else {
#pragma unroll
        for (int i = 0; i < 8; ++i) s[i] *= 0.125f;  // 1/sqrt(64)
      }
      s4[j] = s;
    }

    // ---- online softmax update (rows live across 16-lane halves) ----
    float nm[8], sc[8], ps[8];
#pragma unroll
    for (int i = 0; i < 8; ++i) {
      float v0 = fmaxf(fmaxf(s4[0][i], s4[1][i]), fmaxf(s4[2][i], s4[3][i]));
#pragma unroll
      for (int off = 1; off < 16; off <<= 1) v0 = fmaxf(v0, __shfl_xor(v0, off, 32));
      const float n = fmaxf(mi[i], v0);
      sc[i] = __expf(mi[i] - n);
      nm[i] = n;
      ps[i] = 0.f;
    }
#pragma unroll
    for (int j = 0; j < 4; ++j) {
#pragma unroll
      for (int i = 0; i < 8; ++i) {
        const float p = __expf(s4[j][i] - nm[i]);
        ps[i] += p;
        lP[wave][i + ((lane >> 4) << 3)][j * 16 + (lane & 15)] = (__bf16)p;
      }
    }
#pragma unroll
    for (int i = 0; i < 8; ++i) {
      float t = ps[i];
#pragma unroll
      for (int off = 1; off < 16; off <<= 1) t += __shfl_xor(t, off, 32);
      li[i] = li[i] * sc[i] + t;
      mi[i] = nm[i];
#pragma unroll
      for (int j2 = 0; j2 < 4; ++j2) oacc[j2][i] *= sc[i];
    }

    // ---- O += P @ V  (P re-read from LDS in A-fragment layout) ----
#pragma unroll
    for (int c = 0; c < 2; ++c) {
      const v16bf pa = load_afrag(&lP[wave][lane & 15][c * 32], lane);
#pragma unroll
      for (int j2 = 0; j2 < 4; ++j2) {
        const v16bf vf = *(const v16bf*)&lV[cur][j2 * 16 + (lane & 15)][c * 32 + (lane >> 4) * 16];
        oacc[j2] = wmma_bf16(pa, vf, oacc[j2]);
      }
    }
  }

  // ---- epilogue: normalize, write attn output (t*B+b, e) bf16 ----
#pragma unroll
  for (int i = 0; i < 8; ++i) {
    const float inv = 1.f / li[i];
    const int t = q0 + i + ((lane >> 4) << 3);
#pragma unroll
    for (int j2 = 0; j2 < 4; ++j2) {
      const int d = j2 * 16 + (lane & 15);
      An[(size_t)(t * BATCH + b) * EMB + h * HD + d] = (__bf16)(oacc[j2][i] * inv);
    }
  }
}

// ======================================================================
extern "C" void kernel_launch(void* const* d_in, const int* in_sizes, int n_in,
                              void* d_out, int out_size, void* d_ws, size_t ws_size,
                              hipStream_t stream) {
  const float* q  = (const float*)d_in[0];
  const float* k  = (const float*)d_in[1];
  const float* v  = (const float*)d_in[2];
  const unsigned char* msk = (const unsigned char*)d_in[3];   // jnp.bool_ -> 1 byte
  const float* Wq = (const float*)d_in[4];
  const float* Wk = (const float*)d_in[5];
  const float* Wv = (const float*)d_in[6];
  const float* Wo = (const float*)d_in[7];
  float* out = (float*)d_out;

  // workspace (bf16): Xb 16MB | Wq/Wk/Wv/Wo 2MB each | Qh,Kh,Vt,An 16MB each = 88MB
  char* ws = (char*)d_ws;
  __bf16* Xb  = (__bf16*)(ws);
  __bf16* Wqb = (__bf16*)(ws + (size_t)(16 << 20));
  __bf16* Wkb = (__bf16*)(ws + (size_t)(18 << 20));
  __bf16* Wvb = (__bf16*)(ws + (size_t)(20 << 20));
  __bf16* Wob = (__bf16*)(ws + (size_t)(22 << 20));
  __bf16* Qh  = (__bf16*)(ws + (size_t)(24 << 20));
  __bf16* Kh  = (__bf16*)(ws + (size_t)(40 << 20));
  __bf16* Vt  = (__bf16*)(ws + (size_t)(56 << 20));
  __bf16* An  = (__bf16*)(ws + (size_t)(72 << 20));

  const int nW4 = EMB * EMB / 4;          // 262144
  const int nX4 = MROWS * EMB / 4;        // 2097152
  cvt_bf16_kernel<<<(nW4 + 255) / 256, 256, 0, stream>>>(Wq, Wqb, nW4);
  cvt_bf16_kernel<<<(nW4 + 255) / 256, 256, 0, stream>>>(Wk, Wkb, nW4);
  cvt_bf16_kernel<<<(nW4 + 255) / 256, 256, 0, stream>>>(Wv, Wvb, nW4);
  cvt_bf16_kernel<<<(nW4 + 255) / 256, 256, 0, stream>>>(Wo, Wob, nW4);

  const dim3 gProj(EMB / 128, MROWS / 128);   // (8, 64)
  cvt_bf16_kernel<<<(nX4 + 255) / 256, 256, 0, stream>>>(q, Xb, nX4);
  gemm_kernel<<<gProj, 256, 0, stream>>>(Xb, Wqb, Qh, nullptr, 0);
  cvt_bf16_kernel<<<(nX4 + 255) / 256, 256, 0, stream>>>(k, Xb, nX4);
  gemm_kernel<<<gProj, 256, 0, stream>>>(Xb, Wkb, Kh, nullptr, 0);
  cvt_bf16_kernel<<<(nX4 + 255) / 256, 256, 0, stream>>>(v, Xb, nX4);
  gemm_kernel<<<gProj, 256, 0, stream>>>(Xb, Wvb, Vt, nullptr, 1);

  flash_kernel<<<dim3(T_SEQ / 128, BATCH * HEADS), 256, 0, stream>>>(Qh, Kh, Vt, msk, An);

  gemm_kernel<<<gProj, 256, 0, stream>>>(An, Wob, nullptr, out, 2);
}
